// VectorQuantizer_46797963657238
// MI455X (gfx1250) — compile-verified, matrix operand order checked by reading, not run on hardware
//
#include <hip/hip_runtime.h>
#include <hip/hip_bf16.h>
#include <math.h>

#define M_ROWS 32768
#define K_CODES 8192
#define C_DIM   512

typedef __attribute__((ext_vector_type(16))) __bf16 v16bf;
typedef __attribute__((ext_vector_type(8)))  float  v8f;

union BfFrag {
    v16bf v;
    uint4 q[2];
};

__device__ __forceinline__ unsigned short f32_to_bf16(float f) {
    unsigned int u = __float_as_uint(f);
    u += 0x7FFFu + ((u >> 16) & 1u);   // round-to-nearest-even
    return (unsigned short)(u >> 16);
}
__device__ __forceinline__ float bf16_to_f32(unsigned short h) {
    return __uint_as_float(((unsigned int)h) << 16);
}

__device__ __forceinline__ v8f wmma_bf16(v16bf a, v16bf b, v8f c) {
    return __builtin_amdgcn_wmma_f32_16x16x32_bf16(
        /*neg_a=*/false, a, /*neg_b=*/false, b,
        /*c_mod=*/(short)0, c, /*reuse_a=*/false, /*reuse_b=*/false);
}

// ---------------------------------------------------------------------------
// Kernel 0: zero the accumulation targets (x_sum region in d_out, counts, sse)
// ---------------------------------------------------------------------------
__global__ void vq_zero(float* __restrict__ xsum, int* __restrict__ cnts,
                        float* __restrict__ sse, size_t nx) {
    size_t i = (size_t)blockIdx.x * blockDim.x + threadIdx.x;
    if (i < nx) xsum[i] = 0.0f;
    if (i < K_CODES) cnts[i] = 0;
    if (i == 0) *sse = 0.0f;
}

// ---------------------------------------------------------------------------
// Kernel 1: e = dict / counts; split into bf16 hi/lo; e2[k] = ||e_k||^2
// One wave (32 lanes) per codeword row.
// ---------------------------------------------------------------------------
__global__ __launch_bounds__(256) void vq_prep_e(
    const float* __restrict__ dict, const float* __restrict__ counts,
    unsigned short* __restrict__ e_hi, unsigned short* __restrict__ e_lo,
    float* __restrict__ e2) {
    const int wave = threadIdx.x >> 5;
    const int lane = threadIdx.x & 31;
    const int k = blockIdx.x * 8 + wave;
    const float inv = 1.0f / counts[k];
    const size_t base = (size_t)k * C_DIM;
    float s = 0.0f;
#pragma unroll
    for (int t = 0; t < 16; ++t) {
        const int c = t * 32 + lane;
        const float v = dict[base + c] * inv;
        const unsigned short h = f32_to_bf16(v);
        const float hf = bf16_to_f32(h);
        e_hi[base + c] = h;
        e_lo[base + c] = f32_to_bf16(v - hf);
        s = fmaf(v, v, s);
    }
#pragma unroll
    for (int m = 16; m >= 1; m >>= 1) s += __shfl_xor(s, m, 32);
    if (lane == 0) e2[k] = s;
}

// ---------------------------------------------------------------------------
// Kernel 2: fused distance GEMM + per-row argmin.
// Block = 256 threads (8 waves), owns 32 rows of x, sweeps all K codewords.
// Waves: rtile = wave&1 (two 16-row tiles), cg = wave>>1 (four 32-col groups
// of each 128-col K chunk; each wave does 2 col-tiles of 16).
// s = x_hi*e_hi + x_hi*e_lo + x_lo*e_hi (bf16-split fp32 emulation),
// dist = ||e||^2 - 2 s  (||x||^2 dropped: constant per row).
// ---------------------------------------------------------------------------
__global__ __launch_bounds__(256) void vq_argmin(
    const float* __restrict__ x,
    const unsigned short* __restrict__ e_hi,
    const unsigned short* __restrict__ e_lo,
    const float* __restrict__ e2,
    int* __restrict__ ids) {
    __shared__ alignas(16) unsigned short xs_hi[32 * C_DIM];  // 32 KB
    __shared__ alignas(16) unsigned short xs_lo[32 * C_DIM];  // 32 KB

    const int tid = threadIdx.x;
    const int lane = tid & 31;
    const int wave = tid >> 5;
    const int rtile = wave & 1;   // 0..1
    const int cg = wave >> 1;     // 0..3
    const int row0 = blockIdx.x * 32;

    // Load 32x512 fp32 x-tile, convert to bf16 hi/lo in LDS (done once,
    // reused across all K).
    for (int i = tid; i < 32 * C_DIM; i += 256) {
        const int r = i >> 9;
        const int c = i & (C_DIM - 1);
        const float v = x[(size_t)(row0 + r) * C_DIM + c];
        const unsigned short h = f32_to_bf16(v);
        xs_hi[i] = h;
        xs_lo[i] = f32_to_bf16(v - bf16_to_f32(h));
    }
    __syncthreads();

    float bd[8];
    int bi[8];
#pragma unroll
    for (int j = 0; j < 8; ++j) { bd[j] = 3.4e38f; bi[j] = 0; }

    const int arow = rtile * 16 + (lane & 15);
    const int ac = (lane >> 4) * 16;

    for (int cb = 0; cb < K_CODES; cb += 128) {
        v8f acc0, acc1;
#pragma unroll
        for (int j = 0; j < 8; ++j) { acc0[j] = 0.0f; acc1[j] = 0.0f; }

        const int col0 = cb + cg * 32 + (lane & 15);
        const int col1 = col0 + 16;
        const size_t b0 = (size_t)col0 * C_DIM + ac;
        const size_t b1 = (size_t)col1 * C_DIM + ac;

#pragma unroll 4
        for (int cs = 0; cs < C_DIM; cs += 32) {
            BfFrag ah, al, bh0, bl0, bh1, bl1;
            const uint4* ph = (const uint4*)&xs_hi[arow * C_DIM + cs + ac];
            const uint4* pl = (const uint4*)&xs_lo[arow * C_DIM + cs + ac];
            ah.q[0] = ph[0]; ah.q[1] = ph[1];
            al.q[0] = pl[0]; al.q[1] = pl[1];
            const uint4* g;
            g = (const uint4*)(e_hi + b0 + cs); bh0.q[0] = g[0]; bh0.q[1] = g[1];
            g = (const uint4*)(e_lo + b0 + cs); bl0.q[0] = g[0]; bl0.q[1] = g[1];
            g = (const uint4*)(e_hi + b1 + cs); bh1.q[0] = g[0]; bh1.q[1] = g[1];
            g = (const uint4*)(e_lo + b1 + cs); bl1.q[0] = g[0]; bl1.q[1] = g[1];

            acc0 = wmma_bf16(ah.v, bh0.v, acc0);
            acc0 = wmma_bf16(ah.v, bl0.v, acc0);
            acc0 = wmma_bf16(al.v, bh0.v, acc0);
            acc1 = wmma_bf16(ah.v, bh1.v, acc1);
            acc1 = wmma_bf16(ah.v, bl1.v, acc1);
            acc1 = wmma_bf16(al.v, bh1.v, acc1);
        }

        const float e20 = e2[col0];
        const float e21 = e2[col1];
#pragma unroll
        for (int j = 0; j < 8; ++j) {
            const float d0 = fmaf(-2.0f, acc0[j], e20);
            if (d0 < bd[j]) { bd[j] = d0; bi[j] = col0; }
            const float d1 = fmaf(-2.0f, acc1[j], e21);
            if (d1 < bd[j]) { bd[j] = d1; bi[j] = col1; }
        }
    }

    // Cross-lane argmin within each 16-lane half (cols are striped over lanes).
#pragma unroll
    for (int m = 8; m >= 1; m >>= 1) {
#pragma unroll
        for (int j = 0; j < 8; ++j) {
            const float od = __shfl_xor(bd[j], m, 32);
            const int   oi = __shfl_xor(bi[j], m, 32);
            if (od < bd[j] || (od == bd[j] && oi < bi[j])) { bd[j] = od; bi[j] = oi; }
        }
    }

    // K-loop is done: reuse the x-tile LDS for the per-row reduction arrays.
    __syncthreads();
    float* sd = (float*)xs_hi;  // [4][32]
    int*   si = (int*)xs_lo;    // [4][32]
    if ((lane & 15) == 0) {
        const int rbase = rtile * 16 + (lane >> 4) * 8;
#pragma unroll
        for (int j = 0; j < 8; ++j) {
            sd[cg * 32 + rbase + j] = bd[j];
            si[cg * 32 + rbase + j] = bi[j];
        }
    }
    __syncthreads();
    if (tid < 32) {
        float best = sd[tid];
        int bidx = si[tid];
#pragma unroll
        for (int q = 1; q < 4; ++q) {
            const float d = sd[q * 32 + tid];
            const int i2 = si[q * 32 + tid];
            if (d < best || (d == best && i2 < bidx)) { best = d; bidx = i2; }
        }
        ids[row0 + tid] = bidx;
    }
}

// ---------------------------------------------------------------------------
// Kernel 3: gather x_q = e[ids], write quantized (STE value == x_q),
// scatter counts and x_sum, accumulate sum of squared diffs. 1 block per row.
// ---------------------------------------------------------------------------
__global__ __launch_bounds__(128) void vq_final(
    const float* __restrict__ x, const float* __restrict__ dict,
    const float* __restrict__ counts, const int* __restrict__ ids,
    float* __restrict__ quant, float* __restrict__ xsum,
    int* __restrict__ cnts, float* __restrict__ sse) {
    __shared__ float red[128];
    const int m = blockIdx.x;
    const int tid = threadIdx.x;
    const int id = ids[m];
    const float inv = 1.0f / counts[id];
    float local = 0.0f;
#pragma unroll
    for (int c = tid; c < C_DIM; c += 128) {
        const float xv = x[(size_t)m * C_DIM + c];
        const float ev = dict[(size_t)id * C_DIM + c] * inv;
        quant[(size_t)m * C_DIM + c] = ev;
        const float d = ev - xv;
        local = fmaf(d, d, local);
        atomicAdd(&xsum[(size_t)id * C_DIM + c], xv);
    }
    red[tid] = local;
    __syncthreads();
    for (int s = 64; s >= 1; s >>= 1) {
        if (tid < s) red[tid] += red[tid + s];
        __syncthreads();
    }
    if (tid == 0) {
        atomicAdd(sse, red[0]);
        atomicAdd(&cnts[id], 1);
    }
}

// ---------------------------------------------------------------------------
// Kernel 4: loss, perplexity, counts -> output. Single block.
// ---------------------------------------------------------------------------
__global__ __launch_bounds__(256) void vq_scalars(
    const int* __restrict__ cnts, const float* __restrict__ sse,
    float* __restrict__ out_loss, float* __restrict__ out_ppl,
    float* __restrict__ out_counts) {
    __shared__ float red[256];
    const int tid = threadIdx.x;
    float h = 0.0f;
    for (int k = tid; k < K_CODES; k += 256) {
        const int c = cnts[k];
        out_counts[k] = (float)c;
        const float p = (float)c * (1.0f / (float)M_ROWS);
        h += p * logf(p + 1e-10f);
    }
    red[tid] = h;
    __syncthreads();
    for (int s = 128; s >= 1; s >>= 1) {
        if (tid < s) red[tid] += red[tid + s];
        __syncthreads();
    }
    if (tid == 0) {
        // e_latent_loss == q_latent_loss numerically -> (1 + BETA) * mse
        *out_loss = 1.25f * (*sse) / (float)((size_t)M_ROWS * C_DIM);
        *out_ppl = expf(-red[0]);
    }
}

// ---------------------------------------------------------------------------
extern "C" void kernel_launch(void* const* d_in, const int* in_sizes, int n_in,
                              void* d_out, int out_size, void* d_ws, size_t ws_size,
                              hipStream_t stream) {
    (void)in_sizes; (void)n_in; (void)out_size; (void)ws_size;
    const float* x      = (const float*)d_in[0];  // [M, C]
    const float* dict   = (const float*)d_in[1];  // [K, C]
    const float* counts = (const float*)d_in[2];  // [K]

    // Workspace layout
    unsigned short* e_hi = (unsigned short*)d_ws;                 // K*C bf16
    unsigned short* e_lo = e_hi + (size_t)K_CODES * C_DIM;        // K*C bf16
    float* e2 = (float*)(e_lo + (size_t)K_CODES * C_DIM);         // K f32
    int* ids = (int*)(e2 + K_CODES);                              // M i32
    int* cnts = ids + M_ROWS;                                     // K i32
    float* sse = (float*)(cnts + K_CODES);                        // 1 f32

    // Output layout (flat, reference return order)
    const size_t QN = (size_t)M_ROWS * C_DIM;
    float* out = (float*)d_out;
    float* quant = out;                       // [M*C]
    float* loss = out + QN;                   // [1]
    float* ppl = loss + 1;                    // [1]
    float* ocnt = ppl + 1;                    // [K]
    float* xsum = ocnt + K_CODES;             // [K*C]

    const size_t XN = (size_t)K_CODES * C_DIM;
    vq_zero<<<(unsigned)((XN + 255) / 256), 256, 0, stream>>>(xsum, cnts, sse, XN);
    vq_prep_e<<<K_CODES / 8, 256, 0, stream>>>(dict, counts, e_hi, e_lo, e2);
    vq_argmin<<<M_ROWS / 32, 256, 0, stream>>>(x, e_hi, e_lo, e2, ids);
    vq_final<<<M_ROWS, 128, 0, stream>>>(x, dict, counts, ids, quant, xsum, cnts, sse);
    vq_scalars<<<1, 256, 0, stream>>>(cnts, sse, loss, ppl, ocnt);
}
